// DotProductAttention_15994458211114
// MI455X (gfx1250) — matrix operand order repaired and checked by reading, not compiled, not run on hardware
//
#include <hip/hip_runtime.h>

// ---------------------------------------------------------------------------
// Dot-product attention for MI455X (gfx1250, wave32, WMMA + TDM).
//   in[0] = decoder_hidden  [8, 2048, 1024] f32   (Q)
//   in[1] = encoder_outputs [8, 2048, 1024] f32   (K == V)
//   out   = context [8,2048,1024] f32  ++  attention_weights [8,2048,2048] f32
// Pass 1: S = Q K^T  (TDM double-buffered fp32 tiles -> LDS, cvt bf16, WMMA)
// Pass 2: row softmax in place
// Pass 3: C = P K    (same TDM pipeline; K tile transposed during cvt stage)
// B-fragment loads are software-pipelined one step ahead of the WMMA chain.
// ---------------------------------------------------------------------------

typedef __bf16 bf16_t;
typedef bf16_t bf16x8  __attribute__((ext_vector_type(8)));
typedef bf16_t bf16x16 __attribute__((ext_vector_type(16)));
typedef float  f32x8   __attribute__((ext_vector_type(8)));
typedef unsigned int u32;
typedef u32 u32x4 __attribute__((ext_vector_type(4)));
typedef int i32x4 __attribute__((ext_vector_type(4)));
typedef int i32x8 __attribute__((ext_vector_type(8)));

#define B_SZ 8
#define TQ   2048
#define TK   2048
#define DIM  1024
#define CTX_ELEMS ((size_t)B_SZ * TQ * DIM)

#define TILE 128   // q/k/d tile edge per workgroup
#define KCH  32    // staged reduction depth per iteration (one bf16 WMMA K)
#define STR  40    // padded bf16 LDS row stride: 80B, 16B aligned

union bf8u { bf16_t b[8]; uint4 u; };

static __device__ inline bf16x16 make_frag(uint4 a, uint4 b) {
    bf16x8 lo = __builtin_bit_cast(bf16x8, a);
    bf16x8 hi = __builtin_bit_cast(bf16x8, b);
    return __builtin_shufflevector(lo, hi,
        0,1,2,3,4,5,6,7,8,9,10,11,12,13,14,15);
}

// Load 16 consecutive f32 (global or LDS), convert to bf16, store 32B (16B aligned).
static __device__ inline void cvt16(const float* __restrict__ src,
                                    bf16_t* __restrict__ dst) {
    const float4* s4 = (const float4*)src;
    float4 f0 = s4[0], f1 = s4[1], f2 = s4[2], f3 = s4[3];
    bf8u lo, hi;
    lo.b[0]=(bf16_t)f0.x; lo.b[1]=(bf16_t)f0.y; lo.b[2]=(bf16_t)f0.z; lo.b[3]=(bf16_t)f0.w;
    lo.b[4]=(bf16_t)f1.x; lo.b[5]=(bf16_t)f1.y; lo.b[6]=(bf16_t)f1.z; lo.b[7]=(bf16_t)f1.w;
    hi.b[0]=(bf16_t)f2.x; hi.b[1]=(bf16_t)f2.y; hi.b[2]=(bf16_t)f2.z; hi.b[3]=(bf16_t)f2.w;
    hi.b[4]=(bf16_t)f3.x; hi.b[5]=(bf16_t)f3.y; hi.b[6]=(bf16_t)f3.z; hi.b[7]=(bf16_t)f3.w;
    ((uint4*)dst)[0] = lo.u;
    ((uint4*)dst)[1] = hi.u;
}

static __device__ inline u32 lds_byte_off(const void* p) {
    // generic LDS address: low 32 bits are the LDS byte offset
    return (u32)(unsigned long long)p;
}

// Issue a TDM 2D tile load: tile_h rows x tile_w f32 elems, row stride
// stride_elems, from global to LDS (packed row-major). Tracked by TENSORcnt.
static __device__ inline void tdm_load_2d(u32 lds_off, const void* gptr,
                                          u32 tile_w, u32 tile_h, u32 stride_elems) {
    unsigned long long ga = (unsigned long long)gptr;
    u32x4 g0;
    g0[0] = 1u;                                        // count=1, user descriptor
    g0[1] = lds_off;                                   // lds_addr
    g0[2] = (u32)ga;                                   // global_addr[31:0]
    g0[3] = (u32)((ga >> 32) & 0x01FFFFFFu) | (2u << 30); // ga[56:32] | type=2
    i32x8 g1;
    g1[0] = (int)(2u << 16);                           // data_size = 4B
    g1[1] = (int)((tile_w & 0xFFFFu) << 16);           // tensor_dim0 lo16
    g1[2] = (int)(((tile_w >> 16) & 0xFFFFu) | ((tile_h & 0xFFFFu) << 16));
    g1[3] = (int)(((tile_h >> 16) & 0xFFFFu) | (tile_w << 16)); // tile_dim0
    g1[4] = (int)(tile_h & 0xFFFFu);                   // tile_dim1 (tile_dim2=0)
    g1[5] = (int)stride_elems;                         // tensor_dim0_stride[31:0]
    g1[6] = 0;
    g1[7] = 0;
    i32x4 z4 = {0, 0, 0, 0};
#if defined(__clang_major__) && (__clang_major__ >= 23)
    i32x8 z8 = {0, 0, 0, 0, 0, 0, 0, 0};
    __builtin_amdgcn_tensor_load_to_lds(g0, g1, z4, z4, z8, 0);
#else
    __builtin_amdgcn_tensor_load_to_lds(g0, g1, z4, z4, 0);
#endif
}

// 16q x 128n strip: 8 WMMAs with B-fragment loads pipelined one step ahead,
// so each WMMA waits on loads issued one step earlier (dscnt<=2, not 0).
static __device__ inline void mma_strip(const bf16_t* __restrict__ Bs,
                                        const bf16x16& afrag,
                                        f32x8* __restrict__ acc,
                                        int lr, int hi16) {
    const bf16_t* brow0 = &Bs[lr * STR + hi16 * 8];
    uint4 c0 = ((const uint4*)brow0)[0];
    uint4 c1 = ((const uint4*)brow0)[2];
    #pragma unroll
    for (int j = 0; j < 8; ++j) {
        uint4 n0 = c0, n1 = c1;
        if (j + 1 < 8) {
            const bf16_t* nrow = &Bs[((j + 1) * 16 + lr) * STR + hi16 * 8];
            n0 = ((const uint4*)nrow)[0];
            n1 = ((const uint4*)nrow)[2];
        }
        bf16x16 bfrag = make_frag(c0, c1);
        acc[j] = __builtin_amdgcn_wmma_f32_16x16x32_bf16(
            false, afrag, false, bfrag, (short)0, acc[j], false, false);
        c0 = n0; c1 = n1;
    }
}

// ------------------------- Pass 1: S = Q K^T -------------------------------
__global__ __launch_bounds__(256, 1)
void attn_scores_kernel(const float* __restrict__ Q, const float* __restrict__ K,
                        float* __restrict__ Wout) {
    __shared__ __align__(16) float  Qraw[2][TILE * KCH];   // 2 x 16KB
    __shared__ __align__(16) float  Kraw[2][TILE * KCH];   // 2 x 16KB
    __shared__ __align__(16) bf16_t Qs[TILE * STR];
    __shared__ __align__(16) bf16_t Ks[TILE * STR];

    const int b    = blockIdx.z;
    const int qb   = blockIdx.y * TILE;
    const int kb   = blockIdx.x * TILE;
    const int tid  = threadIdx.x;
    const int lane = tid & 31;
    const int wv   = tid >> 5;       // wave 0..7 -> q rows wv*16..+15
    const int lr   = lane & 15;
    const int hi16 = lane >> 4;

    const float* Qbase = Q + ((size_t)b * TQ + qb) * DIM;
    const float* Kbase = K + ((size_t)b * TK + kb) * DIM;

    const int srow = tid >> 1;        // 0..127
    const int scol = (tid & 1) * 16;  // 0 or 16

    f32x8 acc[8] = {};

    const int NCH = DIM / KCH;        // 32 chunks

    // Prologue: wave 0 starts DMA of chunk 0 into buffer 0.
    if (tid < 32) {
        tdm_load_2d(lds_byte_off(&Qraw[0][0]), Qbase, KCH, TILE, DIM);
        tdm_load_2d(lds_byte_off(&Kraw[0][0]), Kbase, KCH, TILE, DIM);
    }

    for (int i = 0; i < NCH; ++i) {
        const int cur = i & 1;
        if (tid < 32) {
            if (i + 1 < NCH) {
                // prefetch next chunk into the other buffer, then wait for
                // the *current* pair (<=2 outstanding keeps prefetch in flight)
                tdm_load_2d(lds_byte_off(&Qraw[cur ^ 1][0]),
                            Qbase + (size_t)(i + 1) * KCH, KCH, TILE, DIM);
                tdm_load_2d(lds_byte_off(&Kraw[cur ^ 1][0]),
                            Kbase + (size_t)(i + 1) * KCH, KCH, TILE, DIM);
                __builtin_amdgcn_s_wait_tensorcnt(2);
            } else {
                __builtin_amdgcn_s_wait_tensorcnt(0);
            }
        }
        __syncthreads();   // current raw tile visible; prev compute done

        // convert fp32 raw -> padded bf16 tiles
        cvt16(&Qraw[cur][srow * KCH + scol], &Qs[srow * STR + scol]);
        cvt16(&Kraw[cur][srow * KCH + scol], &Ks[srow * STR + scol]);
        __syncthreads();

        // A fragment: lane holds row (wv*16+lr); K elems {0..7,16..23}/{8..15,24..31}
        const bf16_t* arow = &Qs[(wv * 16 + lr) * STR + hi16 * 8];
        uint4 a0 = *(const uint4*)(arow);
        uint4 a1 = *(const uint4*)(arow + 16);
        bf16x16 afrag = make_frag(a0, a1);

        mma_strip(Ks, afrag, acc, lr, hi16);
    }

    #pragma unroll
    for (int j = 0; j < 8; ++j) {
        const int k = kb + j * 16 + lr;
        #pragma unroll
        for (int r = 0; r < 8; ++r) {
            const int q = qb + wv * 16 + r + 8 * hi16;  // C layout: M=r+8*(lane/16)
            Wout[((size_t)b * TQ + q) * TK + k] = acc[j][r];
        }
    }
}

// ------------------------- Pass 2: row softmax -----------------------------
__global__ __launch_bounds__(256)
void attn_softmax_kernel(float* __restrict__ Wbuf) {
    const int lane = threadIdx.x & 31;
    const int wv   = threadIdx.x >> 5;
    const size_t row = (size_t)blockIdx.x * 8 + wv;     // over B*TQ rows
    float4* rowp = (float4*)(Wbuf + row * (size_t)TK);

    float4 v[16];
    float m = -3.4e38f;
    #pragma unroll
    for (int i = 0; i < 16; ++i) {
        v[i] = rowp[i * 32 + lane];
        m = fmaxf(m, fmaxf(fmaxf(v[i].x, v[i].y), fmaxf(v[i].z, v[i].w)));
    }
    #pragma unroll
    for (int off = 16; off > 0; off >>= 1)
        m = fmaxf(m, __shfl_xor(m, off, 32));

    float s = 0.f;
    #pragma unroll
    for (int i = 0; i < 16; ++i) {
        v[i].x = __expf(v[i].x - m); s += v[i].x;
        v[i].y = __expf(v[i].y - m); s += v[i].y;
        v[i].z = __expf(v[i].z - m); s += v[i].z;
        v[i].w = __expf(v[i].w - m); s += v[i].w;
    }
    #pragma unroll
    for (int off = 16; off > 0; off >>= 1)
        s += __shfl_xor(s, off, 32);
    const float inv = 1.0f / s;

    #pragma unroll
    for (int i = 0; i < 16; ++i) {
        v[i].x *= inv; v[i].y *= inv; v[i].z *= inv; v[i].w *= inv;
        rowp[i * 32 + lane] = v[i];
    }
}

// ------------------------- Pass 3: C = P K ---------------------------------
__global__ __launch_bounds__(256, 1)
void attn_context_kernel(const float* __restrict__ P, const float* __restrict__ K,
                         float* __restrict__ Ctx) {
    __shared__ __align__(16) float  Praw[2][TILE * KCH];   // 128q x 32k f32
    __shared__ __align__(16) float  Kraw[2][KCH * TILE];   // 32k x 128d f32
    __shared__ __align__(16) bf16_t Ps[TILE * STR];
    __shared__ __align__(16) bf16_t Kt[TILE * STR];        // transposed: Kt[d][k]

    const int b    = blockIdx.z;
    const int qb   = blockIdx.y * TILE;
    const int db   = blockIdx.x * TILE;
    const int tid  = threadIdx.x;
    const int lane = tid & 31;
    const int wv   = tid >> 5;
    const int lr   = lane & 15;
    const int hi16 = lane >> 4;

    const int prow = tid >> 1;        // 0..127 (q row within tile)
    const int pcol = (tid & 1) * 16;  // 0 or 16 (k within chunk)
    const int kr   = tid >> 3;        // 0..31  (k row within chunk)
    const int dseg = (tid & 7) * 16;  // 0..112 (d segment within tile)

    const float* Pbase = P + ((size_t)b * TQ + qb) * TK;
    const float* Kbase = K + (size_t)b * TK * DIM + db;

    f32x8 acc[8] = {};

    const int NCH = TK / KCH;         // 64 chunks

    if (tid < 32) {
        tdm_load_2d(lds_byte_off(&Praw[0][0]), Pbase, KCH, TILE, TK);
        tdm_load_2d(lds_byte_off(&Kraw[0][0]), Kbase, TILE, KCH, DIM);
    }

    for (int i = 0; i < NCH; ++i) {
        const int cur = i & 1;
        if (tid < 32) {
            if (i + 1 < NCH) {
                tdm_load_2d(lds_byte_off(&Praw[cur ^ 1][0]),
                            Pbase + (size_t)(i + 1) * KCH, KCH, TILE, TK);
                tdm_load_2d(lds_byte_off(&Kraw[cur ^ 1][0]),
                            Kbase + (size_t)(i + 1) * KCH * DIM, TILE, KCH, DIM);
                __builtin_amdgcn_s_wait_tensorcnt(2);
            } else {
                __builtin_amdgcn_s_wait_tensorcnt(0);
            }
        }
        __syncthreads();

        // P chunk: straight convert
        cvt16(&Praw[cur][prow * KCH + pcol], &Ps[prow * STR + pcol]);
        // K chunk: transpose-convert Kraw[k][d] -> Kt[d][k]
        {
            const float4* k4 = (const float4*)&Kraw[cur][kr * TILE + dseg];
            float4 f0 = k4[0], f1 = k4[1], f2 = k4[2], f3 = k4[3];
            float fv[16] = { f0.x,f0.y,f0.z,f0.w, f1.x,f1.y,f1.z,f1.w,
                             f2.x,f2.y,f2.z,f2.w, f3.x,f3.y,f3.z,f3.w };
            #pragma unroll
            for (int t = 0; t < 16; ++t)
                Kt[(dseg + t) * STR + kr] = (bf16_t)fv[t];
        }
        __syncthreads();

        const bf16_t* arow = &Ps[(wv * 16 + lr) * STR + hi16 * 8];
        uint4 a0 = *(const uint4*)(arow);
        uint4 a1 = *(const uint4*)(arow + 16);
        bf16x16 afrag = make_frag(a0, a1);

        mma_strip(Kt, afrag, acc, lr, hi16);
    }

    #pragma unroll
    for (int j = 0; j < 8; ++j) {
        const int d = db + j * 16 + lr;
        #pragma unroll
        for (int r = 0; r < 8; ++r) {
            const int q = qb + wv * 16 + r + 8 * hi16;
            Ctx[((size_t)b * TQ + q) * DIM + d] = acc[j][r];
        }
    }
}

// ---------------------------------------------------------------------------
extern "C" void kernel_launch(void* const* d_in, const int* in_sizes, int n_in,
                              void* d_out, int out_size, void* d_ws, size_t ws_size,
                              hipStream_t stream) {
    (void)in_sizes; (void)n_in; (void)out_size; (void)d_ws; (void)ws_size;
    const float* Q = (const float*)d_in[0];
    const float* K = (const float*)d_in[1];
    float* out  = (float*)d_out;
    float* Ctx  = out;               // [B, Tq, D]
    float* Wbuf = out + CTX_ELEMS;   // [B, Tq, Tk]

    dim3 g1(TK / TILE, TQ / TILE, B_SZ);          // 16 x 16 x 8
    attn_scores_kernel<<<g1, 256, 0, stream>>>(Q, K, Wbuf);

    dim3 g2((B_SZ * TQ) / 8);                     // one wave per row
    attn_softmax_kernel<<<g2, 256, 0, stream>>>(Wbuf);

    dim3 g3(DIM / TILE, TQ / TILE, B_SZ);         // 8 x 16 x 8
    attn_context_kernel<<<g3, 256, 0, stream>>>(Wbuf, K, Ctx);
}
